// FlavorDiffusionModel_34763465294621
// MI455X (gfx1250) — matrix operand order repaired and checked by reading, not compiled, not run on hardware
//
#include <hip/hip_runtime.h>
#include <hip/hip_bf16.h>

// ---------- POD vector types (avoid HIP class-type unions) ----------
typedef float  f32x4  __attribute__((ext_vector_type(4)));
typedef float  v8f    __attribute__((ext_vector_type(8)));
typedef __bf16 v16bf  __attribute__((ext_vector_type(16)));

union FragU  { v16bf v; f32x4 f[2]; };
union Pack16 { f32x4 f[2]; __bf16 h[16]; };

#define GAT_N   25000
#define GAT_NE  200000
#define GAT_ET  (GAT_NE + GAT_N)
#define TILE_K  32
#define LDSPAD  8
#define LDSTR   (TILE_K + LDSPAD)   // 40 bf16 = 80B row stride (16B aligned)
#define BM      256
#define BN      128

// ---------- fp32 -> bf16 convert ----------
__global__ void k_cvt_bf16(const float* __restrict__ s, __bf16* __restrict__ d, long n) {
  long i = (long)blockIdx.x * blockDim.x + threadIdx.x;
  if (i < n) d[i] = (__bf16)s[i];
}

// ---------- fill ----------
__global__ void k_fill(float* __restrict__ p, float v, long n) {
  long i = (long)blockIdx.x * blockDim.x + threadIdx.x;
  if (i < n) p[i] = v;
}

// ---------- WMMA GEMM: C[M,Nd] = A[M,K](bf16) x B[K,Nd](bf16), fp32 accum ----------
// Block tile 256x128, 8 waves, each wave owns a 64x64 tile (4x4 WMMA accumulators).
__global__ __launch_bounds__(256) void k_gemm_wmma(
    const __bf16* __restrict__ A, const __bf16* __restrict__ B,
    float* __restrict__ C, int M, int Nd, int K)
{
  __shared__ __bf16 sA[BM * LDSTR];    // 20 KB
  __shared__ __bf16 sB[BN * LDSTR];    // 10 KB, transposed: sB[n][k]

  const int tid  = threadIdx.x;
  const int lane = tid & 31;
  const int wave = tid >> 5;
  const int l15  = lane & 15;
  const int hi   = lane >> 4;          // 0 | 1 (lane<16 vs lane>=16)
  const int wr   = wave >> 1;          // 0..3 -> 64-row strip
  const int wc   = wave & 1;           // 0..1 -> 64-col strip
  const int blockM = blockIdx.y * BM;
  const int blockN = blockIdx.x * BN;

  v8f acc[4][4] = {};

  // tile-copy mapping: A = one 32-wide row per thread; B = 16 n's of one k-row
  const int gmA  = blockM + tid;
  const int bk   = tid >> 3;           // sB: k 0..31
  const int bn0  = (tid & 7) << 4;     // 16 consecutive n per thread
  const __bf16* aRow = A + (size_t)gmA * K;
  const __bf16* bRow = B + (size_t)bk * Nd + blockN + bn0;

  for (int kb = 0; kb < K; kb += TILE_K) {
    // --- prefetch next chunk (global_prefetch_b8) ---
    if (kb + TILE_K < K) {
      if (gmA < M) __builtin_prefetch(aRow + kb + TILE_K, 0, 1);
      __builtin_prefetch(bRow + (size_t)(kb + TILE_K) * Nd, 0, 1);
    }
    // --- load A tile (row-major), 64B per thread ---
    {
      f32x4 a0 = {}, a1 = {}, a2 = {}, a3 = {};
      if (gmA < M) {
        const f32x4* p = (const f32x4*)(aRow + kb);
        a0 = p[0]; a1 = p[1]; a2 = p[2]; a3 = p[3];
      }
      f32x4* sp = (f32x4*)&sA[tid * LDSTR];
      sp[0] = a0; sp[1] = a1; sp[2] = a2; sp[3] = a3;
    }
    // --- load B tile, transpose into LDS ---
    {
      Pack16 pk;
      const f32x4* p = (const f32x4*)(bRow + (size_t)kb * Nd);
      pk.f[0] = p[0];
      pk.f[1] = p[1];
      #pragma unroll
      for (int i = 0; i < 16; ++i)
        sB[(bn0 + i) * LDSTR + bk] = pk.h[i];
    }
    __syncthreads();

    // --- per-lane fragments per CDNA5 ISA layout ---
    const int ka0 = hi ? 8 : 0;        // A: lanes<16 K{0..7,16..23}; lanes>=16 K{8..15,24..31}
    const int kb0 = hi ? 8 : 0;        // B: lanes<16 K{0..7,16..23}; lanes>=16 K{8..15,24..31}
    FragU fb[4];
    #pragma unroll
    for (int ct = 0; ct < 4; ++ct) {
      int col = wc * 64 + ct * 16 + l15;
      fb[ct].f[0] = *(const f32x4*)&sB[col * LDSTR + kb0];
      fb[ct].f[1] = *(const f32x4*)&sB[col * LDSTR + kb0 + 16];
    }
    #pragma unroll
    for (int rt = 0; rt < 4; ++rt) {
      FragU fa;
      int row = wr * 64 + rt * 16 + l15;
      fa.f[0] = *(const f32x4*)&sA[row * LDSTR + ka0];
      fa.f[1] = *(const f32x4*)&sA[row * LDSTR + ka0 + 16];
      #pragma unroll
      for (int ct = 0; ct < 4; ++ct)
        acc[rt][ct] = __builtin_amdgcn_wmma_f32_16x16x32_bf16(
            false, fa.v, false, fb[ct].v, (short)0, acc[rt][ct], false, false);
    }
    __syncthreads();
  }

  // --- store: VGPR r holds M = base + (hi?8:0) + r, N = l15 (per ISA C/D layout) ---
  const int colBase = blockN + wc * 64 + l15;
  const int rowBase = blockM + wr * 64 + hi * 8;
  if (blockM + BM <= M) {              // fast path: whole tile in range
    #pragma unroll
    for (int rt = 0; rt < 4; ++rt) {
      #pragma unroll
      for (int r = 0; r < 8; ++r) {
        float* rp = C + (size_t)(rowBase + rt * 16 + r) * Nd + colBase;
        #pragma unroll
        for (int ct = 0; ct < 4; ++ct) rp[ct * 16] = acc[rt][ct][r];
      }
    }
  } else {
    #pragma unroll
    for (int rt = 0; rt < 4; ++rt) {
      #pragma unroll
      for (int r = 0; r < 8; ++r) {
        int grow = rowBase + rt * 16 + r;
        if (grow < M) {
          float* rp = C + (size_t)grow * Nd + colBase;
          #pragma unroll
          for (int ct = 0; ct < 4; ++ct) rp[ct * 16] = acc[rt][ct][r];
        }
      }
    }
  }
}

// ---------- per-(node,head) attention coefficients; one wave each ----------
__global__ void k_alpha(const float* __restrict__ H, const float* __restrict__ a_s,
                        const float* __restrict__ a_d, float* __restrict__ als,
                        float* __restrict__ ald, int N, int heads)
{
  long gid = (long)blockIdx.x * blockDim.x + threadIdx.x;
  long wid = gid >> 5;
  int lane = (int)(gid & 31);
  if (wid >= (long)N * heads) return;
  int n  = (int)(wid / heads);
  int hd = (int)(wid % heads);
  const float* hp = H + (size_t)n * heads * 128 + hd * 128;
  const float* as = a_s + hd * 128;
  const float* ad = a_d + hd * 128;
  float s1 = 0.f, s2 = 0.f;
  for (int j = lane; j < 128; j += 32) {
    float hv = hp[j];
    s1 += hv * as[j];
    s2 += hv * ad[j];
  }
  for (int off = 16; off > 0; off >>= 1) {
    s1 += __shfl_xor(s1, off, 32);
    s2 += __shfl_xor(s2, off, 32);
  }
  if (lane == 0) { als[wid] = s1; ald[wid] = s2; }
}

__device__ inline void atomicMaxF(float* addr, float v) {
  if (v >= 0.f) atomicMax((int*)addr, __float_as_int(v));
  else          atomicMin((unsigned int*)addr, __float_as_uint(v));
}

__device__ inline float leaky02(float x) { return x > 0.f ? x : 0.2f * x; }

// ---------- edge pass 1: segment max of leaky_relu logits ----------
__global__ void k_edge_max(const float* __restrict__ als, const float* __restrict__ ald,
                           const int* __restrict__ eidx, float* __restrict__ segmax,
                           int heads)
{
  long gid = (long)blockIdx.x * blockDim.x + threadIdx.x;
  if (gid >= (long)GAT_ET * heads) return;
  int e  = (int)(gid / heads);
  int hd = (int)(gid % heads);
  int s = (e < GAT_NE) ? eidx[e]          : (e - GAT_NE);
  int d = (e < GAT_NE) ? eidx[GAT_NE + e] : (e - GAT_NE);
  float lr = leaky02(als[(size_t)s * heads + hd] + ald[(size_t)d * heads + hd]);
  atomicMaxF(&segmax[(size_t)d * heads + hd], lr);
}

// ---------- edge pass 2: ex = exp(logit - max), segment sum ----------
__global__ void k_edge_exp(const float* __restrict__ als, const float* __restrict__ ald,
                           const int* __restrict__ eidx, const float* __restrict__ segmax,
                           float* __restrict__ exbuf, float* __restrict__ segsum, int heads)
{
  long gid = (long)blockIdx.x * blockDim.x + threadIdx.x;
  if (gid >= (long)GAT_ET * heads) return;
  int e  = (int)(gid / heads);
  int hd = (int)(gid % heads);
  int s = (e < GAT_NE) ? eidx[e]          : (e - GAT_NE);
  int d = (e < GAT_NE) ? eidx[GAT_NE + e] : (e - GAT_NE);
  float lr = leaky02(als[(size_t)s * heads + hd] + ald[(size_t)d * heads + hd]);
  float ex = __expf(lr - segmax[(size_t)d * heads + hd]);
  exbuf[gid] = ex;
  atomicAdd(&segsum[(size_t)d * heads + hd], ex);
}

// ---------- edge pass 3: scatter-add ex * h[src] into Acc[dst]; 4 ch/thread ----------
__global__ void k_edge_scatter(const float* __restrict__ exbuf, const float* __restrict__ H,
                               const int* __restrict__ eidx, float* __restrict__ Acc,
                               int heads, int HC)
{
  long gid = (long)blockIdx.x * blockDim.x + threadIdx.x;
  int q4 = HC >> 2;
  if (gid >= (long)GAT_ET * q4) return;
  int e  = (int)(gid / q4);
  int c0 = ((int)(gid % q4)) << 2;
  int hd = c0 >> 7;
  int s = (e < GAT_NE) ? eidx[e]          : (e - GAT_NE);
  int d = (e < GAT_NE) ? eidx[GAT_NE + e] : (e - GAT_NE);
  float a = exbuf[(size_t)e * heads + hd];
  f32x4 hv = *(const f32x4*)&H[(size_t)s * HC + c0];
  float* ap = &Acc[(size_t)d * HC + c0];
  atomicAdd(ap + 0, a * hv.x);
  atomicAdd(ap + 1, a * hv.y);
  atomicAdd(ap + 2, a * hv.z);
  atomicAdd(ap + 3, a * hv.w);
}

// ---------- finalize: divide by segsum, +bias, relu, emit f32 + bf16 ----------
__global__ void k_finalize(const float* __restrict__ Acc, const float* __restrict__ segsum,
                           const float* __restrict__ bias, float* __restrict__ Hf,
                           __bf16* __restrict__ Hb, int N, int heads, int relu)
{
  int HC = heads * 128;
  long gid = (long)blockIdx.x * blockDim.x + threadIdx.x;
  if (gid >= (long)N * HC) return;
  int n  = (int)(gid / HC);
  int c  = (int)(gid % HC);
  int hd = c >> 7;
  float v = Acc[gid] / segsum[(size_t)n * heads + hd] + bias[c];
  if (relu) v = fmaxf(v, 0.f);
  Hf[gid] = v;
  Hb[gid] = (__bf16)v;
}

// ---------- scorer: gather pairs ----------
__global__ void k_gather_pair(const float* __restrict__ h3, const int* __restrict__ li,
                              const int* __restrict__ ii, float* __restrict__ pair, int B)
{
  long gid = (long)blockIdx.x * blockDim.x + threadIdx.x;
  if (gid >= (long)B * 256) return;
  int b = (int)(gid >> 8);
  int c = (int)(gid & 255);
  int node = (c < 128) ? li[b] : ii[b];
  pair[gid] = h3[(size_t)node * 128 + (c & 127)];
}

// ---------- scorer: dense layer (act 0=relu, 1=none, 2=sigmoid) ----------
__global__ void k_mlp(const float* __restrict__ in, const float* __restrict__ w,
                      const float* __restrict__ bias, float* __restrict__ out,
                      int B, int In, int Out, int act)
{
  long gid = (long)blockIdx.x * blockDim.x + threadIdx.x;
  if (gid >= (long)B * Out) return;
  int b = (int)(gid / Out);
  int j = (int)(gid % Out);
  const float* ip = in + (size_t)b * In;
  float s = bias[j];
  for (int k = 0; k < In; ++k) s += ip[k] * w[(size_t)k * Out + j];
  if (act == 0)      s = fmaxf(s, 0.f);
  else if (act == 2) s = 1.f / (1.f + __expf(-s));
  out[gid] = s;
}

// =====================================================================
extern "C" void kernel_launch(void* const* d_in, const int* in_sizes, int n_in,
                              void* d_out, int out_size, void* d_ws, size_t ws_size,
                              hipStream_t stream) {
  const int NN = GAT_N, B = 4096;

  const float* x    = (const float*)d_in[0];
  const int*   eidx = (const int*)d_in[1];
  const int*   li   = (const int*)d_in[2];
  const int*   ii   = (const int*)d_in[3];
  const float* W1   = (const float*)d_in[4];
  const float* as1  = (const float*)d_in[5];
  const float* ad1  = (const float*)d_in[6];
  const float* b1   = (const float*)d_in[7];
  const float* W2   = (const float*)d_in[8];
  const float* as2  = (const float*)d_in[9];
  const float* ad2  = (const float*)d_in[10];
  const float* b2   = (const float*)d_in[11];
  const float* W3   = (const float*)d_in[12];
  const float* as3  = (const float*)d_in[13];
  const float* ad3  = (const float*)d_in[14];
  const float* b3   = (const float*)d_in[15];
  const float* mw1  = (const float*)d_in[16];
  const float* mb1  = (const float*)d_in[17];
  const float* mw2  = (const float*)d_in[18];
  const float* mb2  = (const float*)d_in[19];
  const float* mw3  = (const float*)d_in[20];
  const float* mb3  = (const float*)d_in[21];
  float* out = (float*)d_out;

  // ---- workspace carve ----
  char* base = (char*)d_ws;
  size_t off = 0;
  auto carve = [&](size_t bytes) -> char* {
    char* p = base + off;
    off = (off + bytes + 255) & ~(size_t)255;
    return p;
  };
  __bf16* Hbf    = (__bf16*)carve((size_t)NN * 1024 * 2);
  __bf16* Wbf    = (__bf16*)carve((size_t)1024 * 1024 * 2);
  float*  H      = (float*) carve((size_t)NN * 1024 * 4);
  float*  Acc    = (float*) carve((size_t)NN * 1024 * 4);
  float*  als    = (float*) carve((size_t)NN * 8 * 4);
  float*  ald    = (float*) carve((size_t)NN * 8 * 4);
  float*  segmax = (float*) carve((size_t)NN * 8 * 4);
  float*  segsum = (float*) carve((size_t)NN * 8 * 4);
  float*  exbuf  = (float*) carve((size_t)GAT_ET * 8 * 4);
  float*  pair   = (float*) carve((size_t)B * 256 * 4);
  float*  z1     = (float*) carve((size_t)B * 128 * 4);
  float*  z2     = (float*) carve((size_t)B * 64 * 4);

  auto blocks = [](long total) -> unsigned { return (unsigned)((total + 255) / 256); };

  // layer-1 input: x -> bf16
  k_cvt_bf16<<<blocks((long)NN * 64), 256, 0, stream>>>(x, Hbf, (long)NN * 64);

  auto run_layer = [&](const float* W, int K, int Nd, const float* a_s, const float* a_d,
                       const float* bias, int heads, int relu) {
    int HC = heads * 128;
    // weights -> bf16
    k_cvt_bf16<<<blocks((long)K * Nd), 256, 0, stream>>>(W, Wbf, (long)K * Nd);
    // H = Hbf @ Wbf   (WMMA)
    dim3 g(Nd / BN, (NN + BM - 1) / BM);
    k_gemm_wmma<<<g, 256, 0, stream>>>(Hbf, Wbf, H, NN, Nd, K);
    // attention coefficients
    k_alpha<<<blocks((long)NN * heads * 32), 256, 0, stream>>>(H, a_s, a_d, als, ald, NN, heads);
    // init segment buffers + accumulator
    k_fill<<<blocks((long)NN * heads), 256, 0, stream>>>(segmax, -3.402823466e38f, (long)NN * heads);
    k_fill<<<blocks((long)NN * heads), 256, 0, stream>>>(segsum, 0.f, (long)NN * heads);
    k_fill<<<blocks((long)NN * HC), 256, 0, stream>>>(Acc, 0.f, (long)NN * HC);
    // segment softmax + aggregate
    k_edge_max    <<<blocks((long)GAT_ET * heads), 256, 0, stream>>>(als, ald, eidx, segmax, heads);
    k_edge_exp    <<<blocks((long)GAT_ET * heads), 256, 0, stream>>>(als, ald, eidx, segmax, exbuf, segsum, heads);
    k_edge_scatter<<<blocks((long)GAT_ET * (HC >> 2)), 256, 0, stream>>>(exbuf, H, eidx, Acc, heads, HC);
    // normalize + bias + relu; write f32 (into H) and bf16 (next layer input)
    k_finalize<<<blocks((long)NN * HC), 256, 0, stream>>>(Acc, segsum, bias, H, Hbf, NN, heads, relu);
  };

  run_layer(W1,   64, 1024, as1, ad1, b1, 8, 1);
  run_layer(W2, 1024, 1024, as2, ad2, b2, 8, 1);
  run_layer(W3, 1024,  128, as3, ad3, b3, 1, 0);

  // scorer
  k_gather_pair<<<blocks((long)B * 256), 256, 0, stream>>>(H, li, ii, pair, B);
  k_mlp<<<blocks((long)B * 128), 256, 0, stream>>>(pair, mw1, mb1, z1, B, 256, 128, 0);
  k_mlp<<<blocks((long)B * 64),  256, 0, stream>>>(z1,   mw2, mb2, z2, B, 128, 64, 0);
  k_mlp<<<blocks((long)B * 1),   256, 0, stream>>>(z2,   mw3, mb3, out, B, 64, 1, 2);
}
